// FlashAttention_4956392259715
// MI455X (gfx1250) — matrix operand order repaired and checked by reading, not compiled, not run on hardware
//
#include <hip/hip_runtime.h>

typedef __attribute__((ext_vector_type(16))) _Float16 v16h;
typedef __attribute__((ext_vector_type(8)))  _Float16 v8h;
typedef __attribute__((ext_vector_type(8)))  float    v8f;
typedef __attribute__((ext_vector_type(4)))  float    v4f;

#define N_SEQ 4096
#define DH    128
#define NPART 8            // split-K partitions for attention

// -------------------- helpers --------------------
__device__ __forceinline__ v16h combine16(v8h lo, v8h hi) {
    v16h r;
#pragma unroll
    for (int i = 0; i < 8; ++i) { r[i] = lo[i]; r[8 + i] = hi[i]; }
    return r;
}

// =====================================================================
// Kernel 1: QKV = x @ W^T + b  (f32 in, f16 Q/K row-major + V transposed out)
// Block tile 128(M) x 128(N), 8 waves, wave tile 64x32 (4x2 WMMA accums)
// Double-buffered LDS: one barrier per k-step, global loads overlap WMMA.
// =====================================================================
#define KSTEP 32
#define LROW  (KSTEP + 8)   // f16 elements per LDS row (40*2=80B rows, 16B aligned)

__device__ __forceinline__ void stage_tile(_Float16* __restrict__ xbuf,
                                           _Float16* __restrict__ wbuf,
                                           int srow, int scol,
                                           const v4f* xv, const v4f* wv) {
    v8h xlo, xhi, wlo, whi;
#pragma unroll
    for (int i = 0; i < 8; ++i) {
        xlo[i] = (_Float16)xv[i >> 2][i & 3];
        wlo[i] = (_Float16)wv[i >> 2][i & 3];
        xhi[i] = (_Float16)xv[2 + (i >> 2)][i & 3];
        whi[i] = (_Float16)wv[2 + (i >> 2)][i & 3];
    }
    *(v8h*)&xbuf[srow * LROW + scol]     = xlo;
    *(v8h*)&xbuf[srow * LROW + scol + 8] = xhi;
    *(v8h*)&wbuf[srow * LROW + scol]     = wlo;
    *(v8h*)&wbuf[srow * LROW + scol + 8] = whi;
}

__global__ __launch_bounds__(256) void qkv_kernel(
    const float* __restrict__ x, const float* __restrict__ W,
    const float* __restrict__ b,
    _Float16* __restrict__ Qh, _Float16* __restrict__ Kh, _Float16* __restrict__ Vt)
{
    __shared__ __align__(16) _Float16 xT[2][128 * LROW];
    __shared__ __align__(16) _Float16 wT[2][128 * LROW];

    const int tid  = threadIdx.x;
    const int lane = tid & 31;
    const int wave = tid >> 5;
    const int wm   = wave & 1;    // 0..1 : 64-row slab
    const int wn   = wave >> 1;   // 0..3 : 32-col slab
    const int m0   = blockIdx.x * 128;
    const int n0   = blockIdx.y * 128;   // 0 / 128 / 256 -> Q / K / V

    const int c  = lane & 15;
    const int hh = lane >> 4;

    v8f acc[4][2];
#pragma unroll
    for (int mt = 0; mt < 4; ++mt)
#pragma unroll
        for (int nt = 0; nt < 2; ++nt) acc[mt][nt] = {};

    // staging: 256 threads cover 128 rows x 32 cols, 16 floats each
    const int srow = tid >> 1;
    const int scol = (tid & 1) * 16;
    const float* xs = x + (size_t)(m0 + srow) * N_SEQ + scol;
    const float* ws = W + (size_t)(n0 + srow) * N_SEQ + scol;

    // prologue: stage k0 = 0 into buffer 0
    {
        v4f xv[4], wv[4];
#pragma unroll
        for (int i = 0; i < 4; ++i) { xv[i] = ((const v4f*)xs)[i]; wv[i] = ((const v4f*)ws)[i]; }
        stage_tile(xT[0], wT[0], srow, scol, xv, wv);
    }
    __syncthreads();

    int buf = 0;
    for (int k0 = 0; k0 < N_SEQ; k0 += KSTEP, buf ^= 1) {
        const bool more = (k0 + KSTEP) < N_SEQ;
        v4f nxv[4], nwv[4];
        if (more) {
            const float* xn = xs + k0 + KSTEP;
            const float* wn = ws + k0 + KSTEP;
#pragma unroll
            for (int i = 0; i < 4; ++i) { nxv[i] = ((const v4f*)xn)[i]; nwv[i] = ((const v4f*)wn)[i]; }
            if (k0 + 2 * KSTEP < N_SEQ) {   // L2 prefetch two steps ahead
                __builtin_prefetch(xs + k0 + 2 * KSTEP, 0, 3);
                __builtin_prefetch(ws + k0 + 2 * KSTEP, 0, 3);
            }
        }

        // A fragments (x): lane holds row M=c, K chunks {8h..8h+7} and {16+8h..+7}
        v16h afr[4];
#pragma unroll
        for (int mt = 0; mt < 4; ++mt) {
            const _Float16* base = &xT[buf][(wm * 64 + mt * 16 + c) * LROW];
            afr[mt] = combine16(*(const v8h*)(base + 8 * hh),
                                *(const v8h*)(base + 16 + 8 * hh));
        }
        // B fragments (W^T): lane holds column N=c, K chunks {8h..8h+7} and {16+8h..+7}
        v16h bfr[2];
#pragma unroll
        for (int nt = 0; nt < 2; ++nt) {
            const _Float16* base = &wT[buf][(wn * 32 + nt * 16 + c) * LROW + 8 * hh];
            bfr[nt] = combine16(*(const v8h*)base, *(const v8h*)(base + 16));
        }
#pragma unroll
        for (int mt = 0; mt < 4; ++mt)
#pragma unroll
            for (int nt = 0; nt < 2; ++nt)
                acc[mt][nt] = __builtin_amdgcn_wmma_f32_16x16x32_f16(
                    false, afr[mt], false, bfr[nt], (short)0, acc[mt][nt], false, false);

        if (more)
            stage_tile(xT[buf ^ 1], wT[buf ^ 1], srow, scol, nxv, nwv);
        __syncthreads();
    }

    // epilogue: bias + route to Q / K / Vt (region uniform per block via n0)
#pragma unroll
    for (int nt = 0; nt < 2; ++nt) {
        const int n    = n0 + wn * 32 + nt * 16 + c;
        const float bn = b[n];
#pragma unroll
        for (int mt = 0; mt < 4; ++mt) {
            const int mbase = m0 + wm * 64 + mt * 16 + 8 * hh;
            if (n0 == 0) {
#pragma unroll
                for (int i = 0; i < 8; ++i)
                    Qh[(size_t)(mbase + i) * DH + n] = (_Float16)(acc[mt][nt][i] + bn);
            } else if (n0 == 128) {
#pragma unroll
                for (int i = 0; i < 8; ++i)
                    Kh[(size_t)(mbase + i) * DH + (n - 128)] = (_Float16)(acc[mt][nt][i] + bn);
            } else {
                v8h pack;
#pragma unroll
                for (int i = 0; i < 8; ++i) pack[i] = (_Float16)(acc[mt][nt][i] + bn);
                *(v8h*)&Vt[(size_t)(n - 256) * N_SEQ + mbase] = pack;  // V stored transposed
            }
        }
    }
}

// =====================================================================
// Kernel 2: split-K flash attention partials.
// grid.y = partition (NPART); each partition covers N_SEQ/NPART keys.
// 4 waves/block, wave owns 16 query rows; key blocks of 64.
// Writes raw O accumulators + per-row (m, l) stats for the combine pass.
// =====================================================================
#define KB    64
#define PROW  72   // 64 + 8 pad (144B rows, 16B aligned)

__global__ __launch_bounds__(128) void attn_kernel(
    const _Float16* __restrict__ Qh, const _Float16* __restrict__ Kh,
    const _Float16* __restrict__ Vt,
    float* __restrict__ Op, float* __restrict__ Ms, float* __restrict__ Ls)
{
    __shared__ __align__(16) _Float16 Pl[4 * 16 * PROW];

    const int tid  = threadIdx.x;
    const int lane = tid & 31;
    const int wave = tid >> 5;
    const int c    = lane & 15;
    const int hh   = lane >> 4;
    const int q0   = blockIdx.x * 64 + wave * 16;
    const int part = blockIdx.y;
    const int jbeg = part * (N_SEQ / NPART);
    const int jend = jbeg + (N_SEQ / NPART);
    _Float16* Pw   = &Pl[wave * 16 * PROW];

    // Q fragments: K=128 -> 4 frags, resident in registers
    v16h qf[4];
#pragma unroll
    for (int kk = 0; kk < 4; ++kk) {
        const _Float16* base = Qh + (size_t)(q0 + c) * DH + kk * 32;
        qf[kk] = combine16(*(const v8h*)(base + 8 * hh),
                           *(const v8h*)(base + 16 + 8 * hh));
    }

    v8f o[8];
#pragma unroll
    for (int nt = 0; nt < 8; ++nt) o[nt] = {};
    float mrow[8], lrow[8];
#pragma unroll
    for (int i = 0; i < 8; ++i) { mrow[i] = -1e30f; lrow[i] = 0.0f; }

    for (int j0 = jbeg; j0 < jend; j0 += KB) {
        if (j0 + KB < jend) {   // pull next key/value block toward WGP$
            __builtin_prefetch(Kh + (size_t)(j0 + KB + c) * DH, 0, 3);
            __builtin_prefetch(Vt + (size_t)(c) * N_SEQ + j0 + KB, 0, 3);
        }
        // ---- S = Q K^T : 16 x 64 tile (4 n-tiles x 4 k-steps) ----
        v8f s[4];
#pragma unroll
        for (int nt = 0; nt < 4; ++nt) {
            v8f a = {};
#pragma unroll
            for (int kk = 0; kk < 4; ++kk) {
                const _Float16* kb = Kh + (size_t)(j0 + nt * 16 + c) * DH + kk * 32 + 8 * hh;
                v16h bf = combine16(*(const v8h*)kb, *(const v8h*)(kb + 16));
                a = __builtin_amdgcn_wmma_f32_16x16x32_f16(
                        false, qf[kk], false, bf, (short)0, a, false, false);
            }
            s[nt] = a;
        }

        // ---- online softmax (rows live across 16-lane halves) ----
        float mnew[8], rs[8];
#pragma unroll
        for (int i = 0; i < 8; ++i) {
            float v = fmaxf(fmaxf(s[0][i], s[1][i]), fmaxf(s[2][i], s[3][i]));
#pragma unroll
            for (int off = 1; off < 16; off <<= 1)
                v = fmaxf(v, __shfl_xor(v, off, 32));
            mnew[i] = fmaxf(mrow[i], v);
        }
#pragma unroll
        for (int i = 0; i < 8; ++i) {
#pragma unroll
            for (int nt = 0; nt < 4; ++nt)
                s[nt][i] = __expf(s[nt][i] - mnew[i]);
            float r = (s[0][i] + s[1][i]) + (s[2][i] + s[3][i]);
#pragma unroll
            for (int off = 1; off < 16; off <<= 1)
                r += __shfl_xor(r, off, 32);
            rs[i] = r;
        }
#pragma unroll
        for (int i = 0; i < 8; ++i) {
            const float f = __expf(mrow[i] - mnew[i]);
            lrow[i] = lrow[i] * f + rs[i];
            mrow[i] = mnew[i];
#pragma unroll
            for (int nt = 0; nt < 8; ++nt) o[nt][i] *= f;
        }

        // ---- P -> LDS (f16), re-fragment into A layout ----
        __syncthreads();
#pragma unroll
        for (int nt = 0; nt < 4; ++nt)
#pragma unroll
            for (int i = 0; i < 8; ++i)
                Pw[(i + 8 * hh) * PROW + nt * 16 + c] = (_Float16)s[nt][i];
        __syncthreads();

        v16h pf[2];
#pragma unroll
        for (int kk = 0; kk < 2; ++kk) {
            const _Float16* base = &Pw[c * PROW + kk * 32];
            pf[kk] = combine16(*(const v8h*)(base + 8 * hh),
                               *(const v8h*)(base + 16 + 8 * hh));
        }

        // ---- O += P @ V  (V stored transposed: contiguous B-frag loads) ----
#pragma unroll
        for (int nt = 0; nt < 8; ++nt) {
#pragma unroll
            for (int kk = 0; kk < 2; ++kk) {
                const _Float16* vb = Vt + (size_t)(nt * 16 + c) * N_SEQ + j0 + kk * 32 + 8 * hh;
                v16h bf = combine16(*(const v8h*)vb, *(const v8h*)(vb + 16));
                o[nt] = __builtin_amdgcn_wmma_f32_16x16x32_f16(
                            false, pf[kk], false, bf, (short)0, o[nt], false, false);
            }
        }
    }

    // ---- epilogue: write raw partial O + per-row stats ----
    float* Ob = Op + (size_t)part * N_SEQ * DH;
#pragma unroll
    for (int i = 0; i < 8; ++i) {
        const int m = q0 + i + 8 * hh;
#pragma unroll
        for (int nt = 0; nt < 8; ++nt)
            Ob[(size_t)m * DH + nt * 16 + c] = o[nt][i];
        if (c == 0) {
            Ms[part * N_SEQ + m] = mrow[i];
            Ls[part * N_SEQ + m] = lrow[i];
        }
    }
}

// =====================================================================
// Kernel 3: combine split-K partials: out = sum_p w_p O_p / sum_p w_p l_p
// =====================================================================
__global__ __launch_bounds__(256) void combine_kernel(
    const float* __restrict__ Op, const float* __restrict__ Ms,
    const float* __restrict__ Ls, float* __restrict__ out)
{
    const int idx = blockIdx.x * 256 + threadIdx.x;   // m*DH + d
    const int m   = idx >> 7;                          // DH = 128
    float M = -1e30f;
#pragma unroll
    for (int p = 0; p < NPART; ++p) M = fmaxf(M, Ms[p * N_SEQ + m]);
    float num = 0.0f, den = 0.0f;
#pragma unroll
    for (int p = 0; p < NPART; ++p) {
        const float w = __expf(Ms[p * N_SEQ + m] - M);
        num += w * Op[(size_t)p * N_SEQ * DH + idx];
        den += w * Ls[p * N_SEQ + m];
    }
    out[idx] = num / den;
}

// =====================================================================
extern "C" void kernel_launch(void* const* d_in, const int* in_sizes, int n_in,
                              void* d_out, int out_size, void* d_ws, size_t ws_size,
                              hipStream_t stream) {
    (void)in_sizes; (void)n_in; (void)out_size; (void)ws_size;
    const float* x = (const float*)d_in[0];
    const float* W = (const float*)d_in[1];
    const float* b = (const float*)d_in[2];
    float* out = (float*)d_out;

    // workspace layout (all 16B aligned)
    _Float16* Qh = (_Float16*)d_ws;                       // [4096,128] f16   (1 MB)
    _Float16* Kh = Qh + (size_t)N_SEQ * DH;               // [4096,128] f16   (1 MB)
    _Float16* Vt = Kh + (size_t)N_SEQ * DH;               // [128,4096] f16   (1 MB)
    float*    Op = (float*)(Vt + (size_t)N_SEQ * DH);     // [NPART,4096,128] (16 MB)
    float*    Ms = Op + (size_t)NPART * N_SEQ * DH;       // [NPART,4096]
    float*    Ls = Ms + (size_t)NPART * N_SEQ;            // [NPART,4096]

    qkv_kernel<<<dim3(32, 3), 256, 0, stream>>>(x, W, b, Qh, Kh, Vt);
    attn_kernel<<<dim3(64, NPART), 128, 0, stream>>>(Qh, Kh, Vt, Op, Ms, Ls);
    combine_kernel<<<dim3((N_SEQ * DH) / 256), 256, 0, stream>>>(Op, Ms, Ls, out);
}
